// CustomizablePatchDominantGradientOrientation_61942018343399
// MI455X (gfx1250) — matrix operand order repaired and checked by reading, not compiled, not run on hardware
//
#include <hip/hip_runtime.h>
#include <math.h>

typedef __attribute__((ext_vector_type(16))) _Float16 v16h;
typedef __attribute__((ext_vector_type(8)))  _Float16 v8h;
typedef __attribute__((ext_vector_type(8)))  float    v8f;

#define PATCH 32
#define HW    1024
#define NBINS 36
#define WAVES_PER_BLOCK 4
#define MAX_BLOCKS 1024

// ---- per-wave LDS region (bytes) ----
#define OFF_BUF0   0          // f32 patch staging A (async dest), 4096 B
#define OFF_BUF1   4096       // f32 patch staging B (async dest), 4096 B
#define OFF_PF16   8192       // f16 patch, column-major [n][k], 2048 B
#define OFF_T1     10240      // T1 = S*P, f16 column-major, 2048 B
#define OFF_T2     12288      // T2 = D*P, f16 column-major, 2048 B
#define OFF_HIST   14336      // 36 f32 bins (pad to 256)
#define OFF_HSM    14592      // 36 f32 smoothed bins (pad to 256)
#define WAVE_REGION 14848
#define SMEM_BYTES (WAVES_PER_BLOCK * WAVE_REGION)

#define PI_F       3.14159265358979323846f
#define HALF_PI_F  1.57079632679489661923f
#define THREE_PI   9.42477796076937971538f
#define BINS_OVER_2PI 5.72957795130823208768f  // 36/(2*pi)

// ---------------- CDNA5 async global->LDS (ASYNCcnt path) ----------------
__device__ __forceinline__ void async_copy_b128(unsigned lds_byte_addr,
                                                const void* sbase,
                                                unsigned voff_bytes) {
    asm volatile("global_load_async_to_lds_b128 %0, %1, %2 offset:0"
                 :
                 : "v"(lds_byte_addr), "v"(voff_bytes), "s"(sbase)
                 : "memory");
}
__device__ __forceinline__ void wait_async_le8() {
    asm volatile("s_wait_asynccnt 8" ::: "memory");
}
__device__ __forceinline__ void wait_async_0() {
    asm volatile("s_wait_asynccnt 0" ::: "memory");
}

// ---------------- closed-form Sobel factor matrices ----------------
__device__ __forceinline__ float Sval(int i, int j) {   // [1,2,1] smooth, edge-replicated
    float v = 0.0f;
    if (j == i)     v = 2.0f;
    if (j == i - 1) v += 1.0f;
    if (j == i + 1) v += 1.0f;
    if (i == 0 && j == 0)                 v = 3.0f;
    if (i == PATCH - 1 && j == PATCH - 1) v = 3.0f;
    return v;
}
__device__ __forceinline__ float Dval(int i, int j) {   // central diff, edge-replicated
    float v = 0.0f;
    if (j == i + 1) v += 1.0f;
    if (j == i - 1) v -= 1.0f;
    if (i == 0 && j == 0)                 v -= 1.0f;
    if (i == PATCH - 1 && j == PATCH - 1) v += 1.0f;
    return v;
}

// ---------------- WMMA fragments (wave32, 16x16x32 f16) ----------------
// A 16x32: lanes 0-15 -> M=lane, K 0..15; lanes 16-31 -> M=lane-16, K 16..31
__device__ __forceinline__ v16h fragA_S(int rowBase, int lane) {
    int m = rowBase + (lane & 15), kofs = (lane & 16) ? 16 : 0;
    v16h a;
#pragma unroll
    for (int h = 0; h < 16; ++h) a[h] = (_Float16)Sval(m, h + kofs);
    return a;
}
__device__ __forceinline__ v16h fragA_D(int rowBase, int lane) {
    int m = rowBase + (lane & 15), kofs = (lane & 16) ? 16 : 0;
    v16h a;
#pragma unroll
    for (int h = 0; h < 16; ++h) a[h] = (_Float16)Dval(m, h + kofs);
    return a;
}
// A operand gathered from a COLUMN-MAJOR f16 LDS tile t[col*32 + row]
__device__ __forceinline__ v16h fragA_cm(const _Float16* t, int rowBase, int lane) {
    int m = rowBase + (lane & 15), kofs = (lane & 16) ? 16 : 0;
    v16h a;
#pragma unroll
    for (int h = 0; h < 16; ++h) {
        int k = h + kofs;
        a[h] = t[k * PATCH + m];
    }
    return a;
}
// B 32x16: lanes 0-15 -> K=0..15, N=lane; lanes 16-31 -> K=16..31, N=lane-16.
// From column-major LDS [n*32+k] this is one contiguous 32B load (2x ds_load_b128).
__device__ __forceinline__ v16h fragB_cm(const _Float16* t, int colBase, int lane) {
    int n = colBase + (lane & 15), kofs = (lane & 16) ? 16 : 0;
    return *(const v16h*)(t + n * PATCH + kofs);
}
__device__ __forceinline__ v16h fragB_Dt(int colBase, int lane) {  // B = D^T
    int n = colBase + (lane & 15), kofs = (lane & 16) ? 16 : 0;
    v16h b;
#pragma unroll
    for (int h = 0; h < 16; ++h) b[h] = (_Float16)Dval(n, kofs + h);
    return b;
}
__device__ __forceinline__ v16h fragB_St(int colBase, int lane) {  // B = S^T (S symmetric)
    int n = colBase + (lane & 15), kofs = (lane & 16) ? 16 : 0;
    v16h b;
#pragma unroll
    for (int h = 0; h < 16; ++h) b[h] = (_Float16)Sval(kofs + h, n);
    return b;
}

__device__ __forceinline__ v8f wmma_f32_16x16x32(v16h a, v16h b) {
    v8f z = {};
    return __builtin_amdgcn_wmma_f32_16x16x32_f16(false, a, false, b, (short)0, z, false, false);
}

// ---------------- fast atan2 (minimax deg-11, err ~2e-6 rad) ----------------
__device__ __forceinline__ float fast_atan2f(float y, float x) {
    float ax = fabsf(x), ay = fabsf(y);
    float mx = fmaxf(ax, ay), mn = fminf(ax, ay);
    float t  = mn * __builtin_amdgcn_rcpf(mx);
    float t2 = t * t;
    float p = fmaf(t2, -0.0117212f, 0.05265332f);
    p = fmaf(t2, p, -0.11643287f);
    p = fmaf(t2, p,  0.19354346f);
    p = fmaf(t2, p, -0.33262347f);
    p = fmaf(t2, p,  0.99997726f);
    float a = t * p;
    if (ay > ax)   a = HALF_PI_F - a;
    if (x < 0.0f)  a = PI_F - a;
    return (y < 0.0f) ? -a : a;
}

// ---------------- kernel: persistent wave32, one patch per wave per iteration ----------------
__global__ void __launch_bounds__(32 * WAVES_PER_BLOCK)
dom_orient_kernel(const float* __restrict__ patch,
                  const float* __restrict__ weight,
                  const float* __restrict__ smooth_w,
                  float* __restrict__ out, int B) {
    extern __shared__ char smem[];
    const int lane = threadIdx.x & 31;
    const int wv   = threadIdx.x >> 5;
    const int wid  = blockIdx.x * WAVES_PER_BLOCK + wv;           // uniform per wave
    const int NW   = gridDim.x * WAVES_PER_BLOCK;

    char* wbase = smem + wv * WAVE_REGION;                        // single shared-derived base
    _Float16* pf16 = (_Float16*)(wbase + OFF_PF16);
    _Float16* t1h  = (_Float16*)(wbase + OFF_T1);
    _Float16* t2h  = (_Float16*)(wbase + OFF_T2);
    float*    hist = (float*)(wbase + OFF_HIST);
    float*    hsm  = (float*)(wbase + OFF_HSM);

    const unsigned dynbase = (unsigned)__builtin_amdgcn_groupstaticsize();
    const unsigned ldsBuf0 = dynbase + (unsigned)wv * WAVE_REGION + OFF_BUF0;

    // ---- hoisted wave-invariant state (lives in VGPRs across all patches) ----
    const int nlo  = lane & 15;
    const int mhi  = (lane & 16) ? 8 : 0;
    v16h aS0 = fragA_S(0, lane),  aS1 = fragA_S(16, lane);
    v16h aD0 = fragA_D(0, lane),  aD1 = fragA_D(16, lane);
    v16h bDt0 = fragB_Dt(0, lane), bDt1 = fragB_Dt(16, lane);
    v16h bSt0 = fragB_St(0, lane), bSt1 = fragB_St(16, lane);

    float wreg[4][8];                        // 0.125 * weight at this lane's 32 pixels
#pragma unroll
    for (int t = 0; t < 4; ++t) {
        int ta = t >> 1, tb = t & 1;
#pragma unroll
        for (int r = 0; r < 8; ++r) {
            int row = 16 * ta + mhi + r, col = 16 * tb + nlo;
            wreg[t][r] = 0.125f * weight[row * PATCH + col];
        }
    }
    const float s0 = smooth_w[0], s1 = smooth_w[1], s2 = smooth_w[2];

    // ---- prime the async pipeline with the first patch ----
    unsigned curOff = 0;                     // byte offset of current staging buffer
    if (wid < B) {
        const char* g0 = (const char*)(patch + (size_t)wid * HW);
#pragma unroll
        for (int i = 0; i < 8; ++i)
            async_copy_b128(ldsBuf0 + (unsigned)i * 512u + (unsigned)lane * 16u,
                            g0, (unsigned)i * 512u + (unsigned)lane * 16u);
    }

    for (int p = wid; p < B; p += NW) {
        // ---- prefetch next patch into the other buffer ----
        const int pn = p + NW;
        const bool havenext = (pn < B);
        if (havenext) {
            const char* gn = (const char*)(patch + (size_t)pn * HW);
            const unsigned nxt = ldsBuf0 + (curOff ^ 4096u);
#pragma unroll
            for (int i = 0; i < 8; ++i)
                async_copy_b128(nxt + (unsigned)i * 512u + (unsigned)lane * 16u,
                                gn, (unsigned)i * 512u + (unsigned)lane * 16u);
        }

        // ---- zero histogram (LDS ops from one wave are in-order) ----
        hist[lane] = 0.0f;                          // bins 0..31
        if (lane < NBINS - 32) hist[32 + lane] = 0.0f;

        // ---- wait for CURRENT buffer (next's 8 may stay outstanding) ----
        if (havenext) wait_async_le8(); else wait_async_0();

        // ---- f32 -> f16, transpose into column-major (lane owns column `lane`) ----
        {
            // offset-select keeps this a single shared-derived pointer -> ds_load, not flat
            const float* src = (const float*)(wbase + curOff);
            _Float16* dstc = pf16 + lane * PATCH;
#pragma unroll
            for (int k0 = 0; k0 < PATCH; k0 += 8) {
                v8h v;
#pragma unroll
                for (int j = 0; j < 8; ++j) v[j] = (_Float16)src[(k0 + j) * PATCH + lane];
                *(v8h*)(dstc + k0) = v;
            }
        }

        // ---- stage 1: T1 = S*P, T2 = D*P (8 WMMAs, K=32) -> column-major f16 ----
        {
            v16h bP0 = fragB_cm(pf16, 0, lane);
            v16h bP1 = fragB_cm(pf16, 16, lane);
#pragma unroll
            for (int ta = 0; ta < 2; ++ta) {
                v16h aS = ta ? aS1 : aS0;
                v16h aD = ta ? aD1 : aD0;
#pragma unroll
                for (int tb = 0; tb < 2; ++tb) {
                    v8f r1 = wmma_f32_16x16x32(aS, tb ? bP1 : bP0);
                    v8f r2 = wmma_f32_16x16x32(aD, tb ? bP1 : bP0);
                    int n = 16 * tb + nlo, mofs = 16 * ta + mhi;
                    v8h o1, o2;
#pragma unroll
                    for (int r = 0; r < 8; ++r) { o1[r] = (_Float16)r1[r]; o2[r] = (_Float16)r2[r]; }
                    *(v8h*)(t1h + n * PATCH + mofs) = o1;     // one ds_store_b128 each
                    *(v8h*)(t2h + n * PATCH + mofs) = o2;
                }
            }
        }

        // ---- stage 2: Gx = T1*D^T, Gy = T2*S^T (8 WMMAs) + fused hist ----
#pragma unroll
        for (int ta = 0; ta < 2; ++ta) {
            v16h a1 = fragA_cm(t1h, 16 * ta, lane);
            v16h a2 = fragA_cm(t2h, 16 * ta, lane);
#pragma unroll
            for (int tb = 0; tb < 2; ++tb) {
                v8f gx = wmma_f32_16x16x32(a1, tb ? bDt1 : bDt0);
                v8f gy = wmma_f32_16x16x32(a2, tb ? bSt1 : bSt0);
                const int t = ta * 2 + tb;
#pragma unroll
                for (int r = 0; r < 8; ++r) {
                    float w  = wreg[t][r];
                    float fx = gx[r] * w;
                    float fy = gy[r] * w;
                    float mag = __builtin_amdgcn_sqrtf(fmaf(fx, fx, fmaf(fy, fy, 1e-18f)));
                    float ori = fast_atan2f(fy, fx + 1e-18f);
                    float obig = fmaf(ori, BINS_OVER_2PI, THREE_PI * BINS_OVER_2PI); // (36,72]
                    float b0f  = floorf(obig);
                    float w1   = obig - b0f;
                    int b0 = (int)b0f - NBINS; if (b0 >= NBINS) b0 -= NBINS;
                    int b1 = b0 + 1;           if (b1 == NBINS) b1 = 0;
                    atomicAdd(&hist[b0], (1.0f - w1) * mag);   // ds_add_f32
                    atomicAdd(&hist[b1], w1 * mag);
                }
            }
        }

        // ---- parallel epilogue: smooth across lanes, butterfly argmax ----
        {
            const float inv_hw = 1.0f / (float)HW;
            int im = (lane == 0) ? NBINS - 1 : lane - 1;
            float h0 = hist[im], h1v = hist[lane], h2 = hist[lane + 1]; // lane+1 <= 32 < 36
            float hsA = (s0 * h0 + s1 * h1v + s2 * h2) * inv_hw;
            hsm[lane] = hsA;
            float bv = hsA; int bi = lane;
            if (lane < NBINS - 32) {                      // lanes 0..3 handle bins 32..35
                int l2 = lane + 32;
                int ip2 = (l2 == NBINS - 1) ? 0 : l2 + 1;
                float g0 = hist[l2 - 1], g1 = hist[l2], g2 = hist[ip2];
                float hsB = (s0 * g0 + s1 * g1 + s2 * g2) * inv_hw;
                hsm[l2] = hsB;
                if (hsB > bv) { bv = hsB; bi = l2; }      // lane index < l2, ties keep lane
            }
#pragma unroll
            for (int off = 16; off > 0; off >>= 1) {
                float ov = __shfl_xor(bv, off, 32);
                int   oi = __shfl_xor(bi, off, 32);
                if (ov > bv || (ov == bv && oi < bi)) { bv = ov; bi = oi; }
            }
            if (lane == 0) {
                int ip = (bi == NBINS - 1) ? 0 : bi + 1;
                int in_ = (bi == 0) ? NBINS - 1 : bi - 1;
                float vp = hsm[ip], vm = hsm[in_];
                float refine = (vp - vm) * 0.5f / (2.0f * bv - (vp + vm));
                float idx = (float)bi + refine;
                out[p] = -(2.0f * PI_F * idx / (float)NBINS - PI_F);
            }
        }

        curOff ^= 4096u;
    }
}

extern "C" void kernel_launch(void* const* d_in, const int* in_sizes, int n_in,
                              void* d_out, int out_size, void* d_ws, size_t ws_size,
                              hipStream_t stream) {
    (void)n_in; (void)out_size; (void)d_ws; (void)ws_size;
    const float* patch    = (const float*)d_in[0];
    const float* weight   = (const float*)d_in[1];
    const float* smooth_w = (const float*)d_in[2];
    float* out = (float*)d_out;

    int B = in_sizes[0] / HW;                       // 32768
    int blocks = (B + WAVES_PER_BLOCK - 1) / WAVES_PER_BLOCK;
    if (blocks > MAX_BLOCKS) blocks = MAX_BLOCKS;   // persistent waves, grid-stride by wave
    dim3 block(32 * WAVES_PER_BLOCK);
    dom_orient_kernel<<<dim3(blocks), block, SMEM_BYTES, stream>>>(patch, weight, smooth_w, out, B);
}
